// GlobalEncoder_40570261078755
// MI455X (gfx1250) — matrix operand order repaired and checked by reading, not compiled, not run on hardware
//
#include <hip/hip_runtime.h>
#include <hip/hip_bf16.h>
#include <math.h>

#define N_TOK  4096
#define DMODEL 512
#define HEADS  8
#define DH     64
#define MLPD   2048
#define GRID_W 80

typedef __attribute__((ext_vector_type(16))) __bf16 v16bf;
typedef __attribute__((ext_vector_type(8)))  float  v8f;
typedef unsigned short ush;

// ---------- helpers ----------
__device__ __forceinline__ ush f2bf(float f) {
  unsigned int u = __float_as_uint(f);
  u += 0x7FFFu + ((u >> 16) & 1u);          // round-to-nearest-even
  return (ush)(u >> 16);
}

__device__ __forceinline__ v8f wmma_bf16(v16bf a, v16bf b, v8f c) {
  return __builtin_amdgcn_wmma_f32_16x16x32_bf16(false, a, false, b, (short)0, c,
                                                 false, false);
}

// A-matrix 16x32 bf16 fragment: lane holds row (lane&15);
// elems j (0-15) -> K = k0 + 16*half + j
__device__ __forceinline__ v16bf load_a_frag(const ush* rowp, int k0, int half) {
  union { v16bf v; uint4 q[2]; } u;
  u.q[0] = *(const uint4*)(rowp + k0 + half * 16);
  u.q[1] = *(const uint4*)(rowp + k0 + half * 16 + 8);
  return u.v;
}

// B-matrix 32x16 bf16 fragment from column-contiguous memory:
// caller passes p = base + (col row)*ld + k0 + half*16 ; elems j -> K=half*16+j
__device__ __forceinline__ v16bf load_b_frag(const ush* p) {
  union { v16bf v; uint4 q[2]; } u;
  u.q[0] = ((const uint4*)p)[0];
  u.q[1] = ((const uint4*)p)[1];
  return u.v;
}

// ---------- weight transpose + fp32->bf16 (Bt[n][k] = W[k][n]) ----------
__global__ void k_wprep(const float* __restrict__ in, ush* __restrict__ out,
                        int K, int Nc) {
  long long id = (long long)blockIdx.x * blockDim.x + threadIdx.x;
  if (id >= (long long)K * Nc) return;
  int n = (int)(id / K), kk = (int)(id % K);
  out[id] = f2bf(in[(size_t)kk * Nc + n]);
}

// ---------- LayerNorm (one block per row, D=512, 256 threads) ----------
__global__ void k_ln(const float* __restrict__ x, const float* __restrict__ g,
                     const float* __restrict__ b, float* __restrict__ outF,
                     ush* __restrict__ outB) {
  __shared__ float s1[256], s2[256];
  int row = blockIdx.x, tid = threadIdx.x;
  const float* xr = x + (size_t)row * DMODEL;
  float a = xr[tid], c = xr[tid + 256];
  s1[tid] = a + c; s2[tid] = a * a + c * c;
  __syncthreads();
  for (int s = 128; s > 0; s >>= 1) {
    if (tid < s) { s1[tid] += s1[tid + s]; s2[tid] += s2[tid + s]; }
    __syncthreads();
  }
  float mean = s1[0] * (1.0f / DMODEL);
  float var  = s2[0] * (1.0f / DMODEL) - mean * mean;
  float rstd = rsqrtf(var + 1e-5f);
  for (int i = tid; i < DMODEL; i += 256) {
    float v = (xr[i] - mean) * rstd * g[i] + b[i];
    if (outF) outF[(size_t)row * DMODEL + i] = v;
    if (outB) outB[(size_t)row * DMODEL + i] = f2bf(v);
  }
}

// ---------- WMMA GEMM: out(MxNc) = A(MxK,bf16) @ Bt(NcxK,bf16)^T + bias (+res)
// one wave per block; wave computes a 32x64 tile:
// 2 A fragments x 4 B fragments -> 8 v_wmma per k-step, B reused across both
// M halves. Plain pointer-increment loop (no branch, no rebasing) so the
// 6 live pointers + 64 acc VGPRs + 48 frag VGPRs stay spill-free. ----------
__global__ void k_gemm(const ush* __restrict__ A, const ush* __restrict__ Bt,
                       const float* __restrict__ bias,
                       const float* __restrict__ residual,
                       float* __restrict__ outF, int M, int K, int Nc) {
  int lane = threadIdx.x;
  int col = lane & 15, half = lane >> 4;
  int m0 = blockIdx.y * 32;
  int n0 = blockIdx.x * 64;
  v8f acc[2][4] = {};
  int ah = half * 8;                               // A-frag element offset
  const ush* pa0 = A + (size_t)(m0 + col) * K;
  const ush* pa1 = pa0 + (size_t)16 * K;
  const ush* pb0 = Bt + (size_t)(n0 + col) * K + half * 16;
  const ush* pb1 = pb0 + (size_t)16 * K;
  const ush* pb2 = pb0 + (size_t)32 * K;
  const ush* pb3 = pb0 + (size_t)48 * K;
  for (int it = K >> 5; it > 0; --it) {
    v16bf a0 = load_a_frag(pa0, 0, half);
    v16bf a1 = load_a_frag(pa1, 0, half);
    v16bf b0 = load_b_frag(pb0);
    v16bf b1 = load_b_frag(pb1);
    v16bf b2 = load_b_frag(pb2);
    v16bf b3 = load_b_frag(pb3);
    acc[0][0] = wmma_bf16(a0, b0, acc[0][0]);
    acc[0][1] = wmma_bf16(a0, b1, acc[0][1]);
    acc[0][2] = wmma_bf16(a0, b2, acc[0][2]);
    acc[0][3] = wmma_bf16(a0, b3, acc[0][3]);
    acc[1][0] = wmma_bf16(a1, b0, acc[1][0]);
    acc[1][1] = wmma_bf16(a1, b1, acc[1][1]);
    acc[1][2] = wmma_bf16(a1, b2, acc[1][2]);
    acc[1][3] = wmma_bf16(a1, b3, acc[1][3]);
    pa0 += 32; pa1 += 32;
    pb0 += 32; pb1 += 32; pb2 += 32; pb3 += 32;
  }
  (void)ah;
  float bia0 = bias[n0 + col],      bia1 = bias[n0 + 16 + col];
  float bia2 = bias[n0 + 32 + col], bia3 = bias[n0 + 48 + col];
#pragma unroll
  for (int mi = 0; mi < 2; ++mi)
#pragma unroll
    for (int r = 0; r < 8; ++r) {
      int mr = m0 + mi * 16 + r + 8 * half;        // C layout: M = r + 8*half
      size_t base = (size_t)mr * Nc + n0 + col;
      float v0 = acc[mi][0][r] + bia0;
      float v1 = acc[mi][1][r] + bia1;
      float v2 = acc[mi][2][r] + bia2;
      float v3 = acc[mi][3][r] + bia3;
      if (residual) {
        v0 += residual[base];      v1 += residual[base + 16];
        v2 += residual[base + 32]; v3 += residual[base + 48];
      }
      outF[base] = v0; outF[base + 16] = v1;
      outF[base + 32] = v2; outF[base + 48] = v3;
    }
}

// ---------- split qkv -> per-head bf16 q (pre-scaled), k, and transposed vT ----
__global__ void k_qkvprep(const float* __restrict__ qkv, ush* __restrict__ q,
                          ush* __restrict__ k, ush* __restrict__ v) {
  int id = blockIdx.x * blockDim.x + threadIdx.x;   // exactly N*512 threads
  int n = id >> 9, c = id & 511;
  int h = c >> 6, d = c & 63;
  const float* r = qkv + (size_t)n * 1536;
  q[((size_t)h * N_TOK + n) * DH + d] = f2bf(r[c] * 0.125f);   // 1/sqrt(64)
  k[((size_t)h * N_TOK + n) * DH + d] = f2bf(r[512 + c]);
  v[((size_t)h * DH + d) * N_TOK + n] = f2bf(r[1024 + c]);
}

// ---------- flash attention: one wave per (head, 16-query tile) ----------
__global__ void k_attn(const ush* __restrict__ q, const ush* __restrict__ kk,
                       const ush* __restrict__ vT, ush* __restrict__ obf) {
  __shared__ __align__(16) ush Pl[16 * 32];
  int lane = threadIdx.x;
  int col = lane & 15, half = lane >> 4;
  int qt = blockIdx.x, h = blockIdx.y;
  const ush* qh = q  + (size_t)h * N_TOK * DH;
  const ush* kh = kk + (size_t)h * N_TOK * DH;
  const ush* vh = vT + (size_t)h * DH * N_TOK;

  const ush* qrow = qh + (size_t)(qt * 16 + col) * DH;
  v16bf aq0 = load_a_frag(qrow, 0, half);
  v16bf aq1 = load_a_frag(qrow, 32, half);

  v8f o0 = {}, o1 = {}, o2 = {}, o3 = {};
  float mrow[8], lrow[8];
#pragma unroll
  for (int r = 0; r < 8; ++r) { mrow[r] = -3.0e38f; lrow[r] = 0.0f; }

  for (int kb = 0; kb < N_TOK; kb += 32) {
    // scores: S(16q x 32keys) = Q @ K^T, reduction over d (two k-steps)
    v8f sa = {}, sb = {};
    sa = wmma_bf16(aq0, load_b_frag(kh + (size_t)(kb + col) * DH + half * 16), sa);
    sa = wmma_bf16(aq1, load_b_frag(kh + (size_t)(kb + col) * DH + 32 + half * 16), sa);
    sb = wmma_bf16(aq0, load_b_frag(kh + (size_t)(kb + 16 + col) * DH + half * 16), sb);
    sb = wmma_bf16(aq1, load_b_frag(kh + (size_t)(kb + 16 + col) * DH + 32 + half * 16), sb);

    float pa[8], pb[8];
#pragma unroll
    for (int r = 0; r < 8; ++r) {
      float t = fmaxf(sa[r], sb[r]);
#pragma unroll
      for (int off = 1; off < 16; off <<= 1) t = fmaxf(t, __shfl_xor(t, off, 32));
      float nm = fmaxf(mrow[r], t);
      float alpha = __expf(mrow[r] - nm);
      pa[r] = __expf(sa[r] - nm);
      pb[r] = __expf(sb[r] - nm);
      float rs = pa[r] + pb[r];
#pragma unroll
      for (int off = 1; off < 16; off <<= 1) rs += __shfl_xor(rs, off, 32);
      lrow[r] = lrow[r] * alpha + rs;
      mrow[r] = nm;
      o0[r] *= alpha; o1[r] *= alpha; o2[r] *= alpha; o3[r] *= alpha;
    }

    // re-layout P (C layout) -> A layout through LDS
    __syncthreads();
#pragma unroll
    for (int r = 0; r < 8; ++r) {
      Pl[(r + 8 * half) * 32 + col]      = f2bf(pa[r]);
      Pl[(r + 8 * half) * 32 + 16 + col] = f2bf(pb[r]);
    }
    __syncthreads();
    v16bf ap = load_a_frag(Pl + col * 32, 0, half);

    // O += P @ V  (vT is (d, key): column-contiguous B)
    o0 = wmma_bf16(ap, load_b_frag(vh + (size_t)(0 * 16 + col) * N_TOK + kb + half * 16), o0);
    o1 = wmma_bf16(ap, load_b_frag(vh + (size_t)(1 * 16 + col) * N_TOK + kb + half * 16), o1);
    o2 = wmma_bf16(ap, load_b_frag(vh + (size_t)(2 * 16 + col) * N_TOK + kb + half * 16), o2);
    o3 = wmma_bf16(ap, load_b_frag(vh + (size_t)(3 * 16 + col) * N_TOK + kb + half * 16), o3);
  }

#pragma unroll
  for (int r = 0; r < 8; ++r) {
    int mr = r + 8 * half;
    float inv = 1.0f / lrow[r];
    size_t base = (size_t)(qt * 16 + mr) * DMODEL + h * DH + col;
    obf[base]      = f2bf(o0[r] * inv);
    obf[base + 16] = f2bf(o1[r] * inv);
    obf[base + 32] = f2bf(o2[r] * inv);
    obf[base + 48] = f2bf(o3[r] * inv);
  }
}

// ---------- GELU (tanh approx, jax.nn.gelu default) ----------
__global__ void k_gelu(const float* __restrict__ u, ush* __restrict__ g) {
  size_t id = (size_t)blockIdx.x * blockDim.x + threadIdx.x;
  float x = u[id];
  float t = 0.7978845608028654f * (x + 0.044715f * x * x * x);
  g[id] = f2bf(0.5f * x * (1.0f + tanhf(t)));
}

// ---------- PEG helpers ----------
__global__ void k_posmin(const int* __restrict__ pos, int* __restrict__ pmin) {
  int n = blockIdx.x * blockDim.x + threadIdx.x;
  if (n < N_TOK) {
    atomicMin(&pmin[0], pos[2 * n]);
    atomicMin(&pmin[1], pos[2 * n + 1]);
  }
}

__global__ void k_scatter(const float* __restrict__ x, const int* __restrict__ pos,
                          const int* __restrict__ pmin, float* __restrict__ grid) {
  int id = blockIdx.x * blockDim.x + threadIdx.x;   // N*512 exact
  int n = id >> 9, c = id & 511;
  int p0 = pos[2 * n] - pmin[0];
  int p1 = pos[2 * n + 1] - pmin[1];
  grid[((size_t)p0 * GRID_W + p1) * DMODEL + c] = x[id];
}

__global__ void k_peg(const float* __restrict__ grid, const int* __restrict__ pos,
                      const int* __restrict__ pmin, const float* __restrict__ cw,
                      const float* __restrict__ cb, float* __restrict__ out) {
  int id = blockIdx.x * blockDim.x + threadIdx.x;   // N*512 exact
  int n = id >> 9, c = id & 511;
  int p0 = pos[2 * n] - pmin[0];
  int p1 = pos[2 * n + 1] - pmin[1];
  float acc = grid[((size_t)p0 * GRID_W + p1) * DMODEL + c] + cb[c];
#pragma unroll
  for (int ky = 0; ky < 3; ++ky)
#pragma unroll
    for (int kx = 0; kx < 3; ++kx) {
      int gx = p0 + kx - 1, gy = p1 + ky - 1;
      if (gx >= 0 && gx < GRID_W && gy >= 0 && gy < GRID_W)
        acc += cw[c * 9 + ky * 3 + kx] *
               grid[((size_t)gx * GRID_W + gy) * DMODEL + c];
    }
  out[id] = acc;
}

// ---------- host ----------
extern "C" void kernel_launch(void* const* d_in, const int* in_sizes, int n_in,
                              void* d_out, int out_size, void* d_ws, size_t ws_size,
                              hipStream_t stream) {
  (void)in_sizes; (void)n_in; (void)out_size; (void)ws_size;
  const float* x     = (const float*)d_in[0];
  const int*   pos   = (const int*)d_in[1];
  const float* Wqkv  = (const float*)d_in[2];
  const float* bqkv  = (const float*)d_in[3];
  const float* Wo    = (const float*)d_in[4];
  const float* bo    = (const float*)d_in[5];
  const float* ln1g  = (const float*)d_in[6];
  const float* ln1b  = (const float*)d_in[7];
  const float* W1    = (const float*)d_in[8];
  const float* b1    = (const float*)d_in[9];
  const float* W2    = (const float*)d_in[10];
  const float* b2    = (const float*)d_in[11];
  const float* ln2g  = (const float*)d_in[12];
  const float* ln2b  = (const float*)d_in[13];
  const float* convw = (const float*)d_in[14];
  const float* convb = (const float*)d_in[15];
  const float* ng    = (const float*)d_in[16];
  const float* nb    = (const float*)d_in[17];

  char* p = (char*)d_ws;
  auto alloc = [&](size_t bytes) -> char* {
    char* r = p; p += (bytes + 255) & ~(size_t)255; return r;
  };
  ush*   WqkvT = (ush*)alloc((size_t)2 * 1536 * 512 * 2);
  ush*   WoT   = (ush*)alloc((size_t)2 * 512 * 512 * 2);
  ush*   W1T   = (ush*)alloc((size_t)2 * 2048 * 512 * 2);
  ush*   W2T   = (ush*)alloc((size_t)2 * 512 * 2048 * 2);
  ush*   lnbf  = (ush*)alloc((size_t)N_TOK * DMODEL * 2);
  float* big   = (float*)alloc((size_t)N_TOK * MLPD * 4);   // qkv_f then mlp hidden
  ush*   qbf   = (ush*)alloc((size_t)N_TOK * DMODEL * 2);
  ush*   kbf   = (ush*)alloc((size_t)N_TOK * DMODEL * 2);
  ush*   vbf   = (ush*)alloc((size_t)N_TOK * DMODEL * 2);
  ush*   obf   = (ush*)alloc((size_t)N_TOK * DMODEL * 2);
  ush*   gbf   = (ush*)alloc((size_t)N_TOK * MLPD * 2);
  float* xa    = (float*)alloc((size_t)N_TOK * DMODEL * 4);
  float* xb    = (float*)alloc((size_t)N_TOK * DMODEL * 4);
  float* gridb = (float*)alloc((size_t)GRID_W * GRID_W * DMODEL * 4);
  int*   pmin  = (int*)alloc(256);

  // ---- weight prep (transpose + bf16) ----
  for (int l = 0; l < 2; ++l) {
    k_wprep<<<(1536 * 512 + 255) / 256, 256, 0, stream>>>(
        Wqkv + (size_t)l * 512 * 1536, WqkvT + (size_t)l * 1536 * 512, 512, 1536);
    k_wprep<<<(512 * 512 + 255) / 256, 256, 0, stream>>>(
        Wo + (size_t)l * 512 * 512, WoT + (size_t)l * 512 * 512, 512, 512);
    k_wprep<<<(2048 * 512 + 255) / 256, 256, 0, stream>>>(
        W1 + (size_t)l * 512 * 2048, W1T + (size_t)l * 2048 * 512, 512, 2048);
    k_wprep<<<(512 * 2048 + 255) / 256, 256, 0, stream>>>(
        W2 + (size_t)l * 2048 * 512, W2T + (size_t)l * 512 * 2048, 2048, 512);
  }

  const int ELT_BLK = N_TOK * DMODEL / 256;        // 8192
  auto block = [&](int l, const float* xin, float* xmid, float* xout) {
    k_ln<<<N_TOK, 256, 0, stream>>>(xin, ln1g + l * 512, ln1b + l * 512,
                                    nullptr, lnbf);
    k_gemm<<<dim3(1536 / 64, N_TOK / 32), 32, 0, stream>>>(
        lnbf, WqkvT + (size_t)l * 1536 * 512, bqkv + l * 1536, nullptr, big,
        N_TOK, 512, 1536);
    k_qkvprep<<<ELT_BLK, 256, 0, stream>>>(big, qbf, kbf, vbf);
    k_attn<<<dim3(N_TOK / 16, HEADS), 32, 0, stream>>>(qbf, kbf, vbf, obf);
    k_gemm<<<dim3(512 / 64, N_TOK / 32), 32, 0, stream>>>(
        obf, WoT + (size_t)l * 512 * 512, bo + l * 512, xin, xmid,
        N_TOK, 512, 512);
    k_ln<<<N_TOK, 256, 0, stream>>>(xmid, ln2g + l * 512, ln2b + l * 512,
                                    nullptr, lnbf);
    k_gemm<<<dim3(2048 / 64, N_TOK / 32), 32, 0, stream>>>(
        lnbf, W1T + (size_t)l * 2048 * 512, b1 + l * 2048, nullptr, big,
        N_TOK, 512, 2048);
    k_gelu<<<N_TOK * MLPD / 256, 256, 0, stream>>>(big, gbf);
    k_gemm<<<dim3(512 / 64, N_TOK / 32), 32, 0, stream>>>(
        gbf, W2T + (size_t)l * 512 * 2048, b2 + l * 512, xmid, xout,
        N_TOK, 2048, 512);
  };

  // layer 0
  block(0, x, xa, xb);

  // PEG (depthwise 3x3 conv on sparse grid)
  hipMemsetAsync(pmin, 0x7f, 8, stream);
  hipMemsetAsync(gridb, 0, (size_t)GRID_W * GRID_W * DMODEL * 4, stream);
  k_posmin<<<(N_TOK + 255) / 256, 256, 0, stream>>>(pos, pmin);
  k_scatter<<<ELT_BLK, 256, 0, stream>>>(xb, pos, pmin, gridb);
  k_peg<<<ELT_BLK, 256, 0, stream>>>(gridb, pos, pmin, convw, convb, xa);

  // layer 1
  block(1, xa, xb, xa);

  // final LayerNorm -> d_out (fp32)
  k_ln<<<N_TOK, 256, 0, stream>>>(xa, ng, nb, (float*)d_out, nullptr);
}
